// Model_73151882985929
// MI455X (gfx1250) — matrix operand order repaired and checked by reading, not compiled, not run on hardware
//
#include <hip/hip_runtime.h>
#include <hip/hip_bf16.h>
#include <math.h>

// ---------------------------------------------------------------------------
// CDNA5 (gfx1250, wave32) WMMA helpers
// ---------------------------------------------------------------------------
typedef __attribute__((ext_vector_type(16))) __bf16 v16bf;
typedef __attribute__((ext_vector_type(8)))  float  v8f;

union Frag16 {
    v16bf v;
    uint4 q[2];
    unsigned short us[16];
};

__device__ __forceinline__ v8f wmma_bf16(v16bf a, v16bf b, v8f c) {
    // D = A(16x32 bf16) * B(32x16 bf16) + C(16x16 f32)
    return __builtin_amdgcn_wmma_f32_16x16x32_bf16(
        /*neg_a=*/false, a, /*neg_b=*/false, b,
        /*c_mod=*/(short)0, c, /*reuse_a=*/false, /*reuse_b=*/false);
}

__device__ __forceinline__ unsigned short f32_to_bf16(float x) {
    unsigned u = __float_as_uint(x);
    unsigned r = u + 0x7FFFu + ((u >> 16) & 1u);  // round-to-nearest-even
    return (unsigned short)(r >> 16);
}

__device__ __forceinline__ float sigf(float x) {
    return 1.0f / (1.0f + __expf(-x));
}

// B fragment (K=32 x N=16, bf16) from a row-major [512, kipad] weight matrix.
// element e: n = n0 + (lane&15); e<8: k = kc*32 + (lane>>4)*8 + e ; e>=8: k = kc*32 + 16 + (lane>>4)*8 + (e-8)
__device__ __forceinline__ v16bf load_b_frag(const unsigned short* __restrict__ W,
                                             int n0, int kipad, int kc, int lane) {
    const int l = lane & 15, hi = lane >> 4;
    const unsigned short* p = W + (size_t)(n0 + l) * kipad + kc * 32 + hi * 8;
    Frag16 f;
    f.q[0] = *(const uint4*)(p);
    f.q[1] = *(const uint4*)(p + 16);
    return f.v;
}

// A fragment (M=16 x K=32, bf16) from row-major activations, row stride `stride`.
// lane l(<16)/l+16 holds row m=l; e<8: k=kc*32+hi*8+e ; e>=8: k=kc*32+16+hi*8+(e-8)
__device__ __forceinline__ v16bf load_a_frag_g(const unsigned short* __restrict__ X,
                                               size_t row0, int stride, int kc, int lane) {
    const int l = lane & 15, hi = lane >> 4;
    const unsigned short* p = X + (row0 + l) * (size_t)stride + kc * 32 + hi * 8;
    Frag16 f;
    f.q[0] = *(const uint4*)(p);
    f.q[1] = *(const uint4*)(p + 16);
    return f.v;
}

__device__ __forceinline__ v16bf load_a_frag_lds(const unsigned short* H, int kc, int lane) {
    const int l = lane & 15, hi = lane >> 4;
    const unsigned short* p = H + l * 128 + kc * 32 + hi * 8;
    Frag16 f;
    f.q[0] = *(const uint4*)(p);
    f.q[1] = *(const uint4*)(p + 16);
    return f.v;
}

// ---------------------------------------------------------------------------
// Problem constants
// ---------------------------------------------------------------------------
#define NN   512
#define CC   3
#define TT   300
#define VV   15
#define MM   2
#define HH   128
#define BNM  1024          // N*M
#define ROWS 307200        // T * BNM
#define NC   60

// Workspace layout (bytes, 128B aligned)
#define OFF_LHAT  ((size_t)0)                       // 225 f32
#define OFF_B0    ((size_t)1024)                    // 512 f32
#define OFF_B1    ((size_t)3072)                    // 512 f32
#define OFF_WIH0  ((size_t)8192)                    // 512x64  bf16 (padded from 45)
#define OFF_WHH0  (OFF_WIH0 + (size_t)512*64*2)     // 512x128 bf16
#define OFF_WIH1  (OFF_WHH0 + (size_t)512*128*2)    // 512x128 bf16
#define OFF_WHH1  (OFF_WIH1 + (size_t)512*128*2)    // 512x128 bf16
#define OFF_X0    (OFF_WHH1 + (size_t)512*128*2)    // ROWS x 64 bf16 (cheb out, padded)
#define OFF_HS0   (OFF_X0  + (size_t)ROWS*64*2)     // ROWS x 128 bf16 (layer0 hidden)
#define OFF_HLAST (OFF_HS0 + (size_t)ROWS*128*2)    // 1024 x 128 f32

// ---------------------------------------------------------------------------
// Init: build normalized Laplacian L_hat = -D^-1/2 A D^-1/2 on device
// ---------------------------------------------------------------------------
__global__ __launch_bounds__(256) void build_lhat_kernel(float* __restrict__ lhat) {
    __shared__ float A[225];
    __shared__ float dinv[15];
    const int tid = threadIdx.x;
    const unsigned char SRC[28] = {0,2,4,1,3,5,6,8,10,7,9,11,13,12,2,4,14,3,5,14,8,10,13,9,11,13,12,13};
    const unsigned char DST[28] = {2,4,14,3,5,14,8,10,13,9,11,13,14,13,0,2,4,1,3,5,6,8,10,7,9,11,13,14};
    if (tid < 225) A[tid] = 0.f;
    __syncthreads();
    if (tid < 28) A[(int)DST[tid] * 15 + (int)SRC[tid]] = 1.f;  // benign dup writes of 1.0
    __syncthreads();
    if (tid < 15) {
        float d = 0.f;
        for (int j = 0; j < 15; ++j) d += A[tid * 15 + j];
        dinv[tid] = d > 0.f ? rsqrtf(d) : 0.f;
    }
    __syncthreads();
    if (tid < 225) {
        int v = tid / 15, j = tid % 15;
        lhat[tid] = -(dinv[v] * A[tid] * dinv[j]);
    }
}

__global__ __launch_bounds__(256) void pad_convert_kernel(unsigned short* __restrict__ dst,
                                                          const float* __restrict__ src,
                                                          int rows, int sc, int dc) {
    int i = blockIdx.x * 256 + threadIdx.x;
    if (i >= rows * dc) return;
    int r = i / dc, c = i % dc;
    dst[i] = (c < sc) ? f32_to_bf16(src[r * sc + c]) : (unsigned short)0;
}

__global__ __launch_bounds__(256) void bias_combine_kernel(float* __restrict__ dst,
                                                           const float* __restrict__ a,
                                                           const float* __restrict__ b, int n) {
    int i = blockIdx.x * 256 + threadIdx.x;
    if (i < n) dst[i] = a[i] + b[i];
}

// ---------------------------------------------------------------------------
// Fused permute + ChebConv(K=2) + bf16 pack into [t*1024 + nm][64]
// ---------------------------------------------------------------------------
__global__ __launch_bounds__(256)
void cheb_kernel(const float* __restrict__ x1,
                 const float* __restrict__ w0, const float* __restrict__ w1,
                 const float* __restrict__ cb, const float* __restrict__ lhat,
                 unsigned short* __restrict__ X) {
    __shared__ float sl[225];
    __shared__ float sw0[9], sw1[9], sb[3];
    const int tid = threadIdx.x;
    if (tid < 225) sl[tid] = lhat[tid];
    if (tid < 9) { sw0[tid] = w0[tid]; sw1[tid] = w1[tid]; }
    if (tid < 3) sb[tid] = cb[tid];
    __syncthreads();

    const int b = blockIdx.x * 256 + tid;        // 0..ROWS-1
    const int n   = b / (MM * TT);
    const int rem = b % (MM * TT);
    const int m   = rem / TT;
    const int t   = rem % TT;

    float x[VV][CC];
    #pragma unroll
    for (int c = 0; c < CC; ++c) {
        size_t base = (((size_t)n * CC + c) * TT + t) * (VV * MM) + m;
        #pragma unroll
        for (int v = 0; v < VV; ++v) x[v][c] = x1[base + v * MM];
    }
    float tx[VV][CC];
    #pragma unroll
    for (int v = 0; v < VV; ++v) {
        float a0 = 0.f, a1 = 0.f, a2 = 0.f;
        #pragma unroll
        for (int j = 0; j < VV; ++j) {
            float l = sl[v * 15 + j];
            a0 += l * x[j][0]; a1 += l * x[j][1]; a2 += l * x[j][2];
        }
        tx[v][0] = a0; tx[v][1] = a1; tx[v][2] = a2;
    }
    union { unsigned short us[64]; uint4 q[8]; } out;
    #pragma unroll
    for (int v = 0; v < VV; ++v)
        #pragma unroll
        for (int o = 0; o < CC; ++o) {
            float y = sb[o];
            #pragma unroll
            for (int c = 0; c < CC; ++c)
                y += x[v][c] * sw0[o * 3 + c] + tx[v][c] * sw1[o * 3 + c];
            out.us[v * 3 + o] = f32_to_bf16(y);
        }
    #pragma unroll
    for (int k = VV * CC; k < 64; ++k) out.us[k] = 0;

    const size_t r = (size_t)t * BNM + (n * MM + m);
    uint4* dst = (uint4*)(X + r * 64);
    #pragma unroll
    for (int i = 0; i < 8; ++i) dst[i] = out.q[i];
}

// ---------------------------------------------------------------------------
// LSTM layer: 64 WGs x 16 batch rows, full time loop per WG.
// Wave w owns h-columns [w*16, w*16+16) for all four gates (i/f/g/o), so the
// cell update is wave-local; h is exchanged via a double-buffered 4KB LDS tile.
// Whh fragments are register-resident; Wih streams from (L2-hot) global.
// ---------------------------------------------------------------------------
template <int KIPAD>
__global__ __launch_bounds__(256)
void lstm_layer_kernel(const unsigned short* __restrict__ Xin,  // [T*1024, KIPAD] bf16
                       const unsigned short* __restrict__ Wih,  // [512, KIPAD] bf16
                       const unsigned short* __restrict__ Whh,  // [512, 128]  bf16
                       const float* __restrict__ bias,          // [512] (bih+bhh)
                       unsigned short* __restrict__ hs_out,     // [T*1024,128] bf16 or null
                       float* __restrict__ hlast,               // [1024,128] f32 or null
                       int T) {
    constexpr int KC_I = KIPAD / 32;
    __shared__ __align__(16) unsigned short hbuf[2][16][128];

    const int tid  = threadIdx.x;
    const int wave = tid >> 5;
    const int lane = tid & 31;
    const int l    = lane & 15;
    const int hi   = lane >> 4;
    const int nm0  = blockIdx.x * 16;

    for (int i = tid; i < 16 * 128; i += 256) (&hbuf[0][0][0])[i] = 0;

    // Register-resident recurrent weight fragments: [kchunk][gate]
    v16bf whh[4][4];
    #pragma unroll
    for (int kc = 0; kc < 4; ++kc)
        #pragma unroll
        for (int g = 0; g < 4; ++g)
            whh[kc][g] = load_b_frag(Whh, g * 128 + wave * 16, 128, kc, lane);

    float bval[4];
    #pragma unroll
    for (int g = 0; g < 4; ++g) bval[g] = bias[g * 128 + wave * 16 + l];

    float cstate[8];
    #pragma unroll
    for (int e = 0; e < 8; ++e) cstate[e] = 0.f;

    __syncthreads();

    for (int t = 0; t < T; ++t) {
        const int cur = t & 1, nxt = cur ^ 1;
        const size_t xrow = (size_t)t * BNM + nm0;

        v8f acc[4];
        #pragma unroll
        for (int g = 0; g < 4; ++g) {
            v8f z;
            #pragma unroll
            for (int e = 0; e < 8; ++e) z[e] = bval[g];
            acc[g] = z;
        }

        // Input-to-hidden contribution (streams Wih frags from L2)
        #pragma unroll
        for (int kc = 0; kc < KC_I; ++kc) {
            v16bf a = load_a_frag_g(Xin, xrow, KIPAD, kc, lane);
            #pragma unroll
            for (int g = 0; g < 4; ++g) {
                v16bf bf = load_b_frag(Wih, g * 128 + wave * 16, KIPAD, kc, lane);
                acc[g] = wmma_bf16(a, bf, acc[g]);
            }
        }

        // Prefetch next timestep's activation rows (global_prefetch_b8)
        if (t + 1 < T)
            __builtin_prefetch(Xin + (xrow + BNM + l) * KIPAD, 0, 1);

        // Recurrent contribution from the LDS h tile
        #pragma unroll
        for (int kc = 0; kc < 4; ++kc) {
            v16bf a = load_a_frag_lds(&hbuf[cur][0][0], kc, lane);
            #pragma unroll
            for (int g = 0; g < 4; ++g)
                acc[g] = wmma_bf16(a, whh[kc][g], acc[g]);
        }

        // Cell update; D-frag element e -> (m = e + 8*hi, j = wave*16 + l)
        #pragma unroll
        for (int e = 0; e < 8; ++e) {
            float iv = sigf(acc[0][e]);
            float fv = sigf(acc[1][e]);
            float gv = tanhf(acc[2][e]);
            float ov = sigf(acc[3][e]);
            float cv = fv * cstate[e] + iv * gv;
            cstate[e] = cv;
            float hv = ov * tanhf(cv);
            const int m = e + 8 * hi;
            const int j = wave * 16 + l;
            unsigned short hb = f32_to_bf16(hv);
            hbuf[nxt][m][j] = hb;
            if (hs_out) hs_out[(xrow + m) * 128 + j] = hb;
            if (hlast && t == T - 1) hlast[(size_t)(nm0 + m) * 128 + j] = hv;
        }
        __syncthreads();
    }
}

// ---------------------------------------------------------------------------
// FC(128 -> 60) + softmax, one row per 64-thread block
// ---------------------------------------------------------------------------
__global__ __launch_bounds__(64)
void fc_softmax_kernel(const float* __restrict__ hlast, const float* __restrict__ fcw,
                       const float* __restrict__ fcb, float* __restrict__ out) {
    __shared__ float hrow[128];
    __shared__ float red[64];
    const int row = blockIdx.x;
    const int tid = threadIdx.x;
    hrow[tid]      = hlast[(size_t)row * 128 + tid];
    hrow[tid + 64] = hlast[(size_t)row * 128 + 64 + tid];
    __syncthreads();

    float logit = 0.f, v = -INFINITY;
    if (tid < NC) {
        float s = fcb[tid];
        #pragma unroll 8
        for (int k = 0; k < 128; ++k) s += hrow[k] * fcw[tid * 128 + k];
        logit = s; v = s;
    }
    red[tid] = v; __syncthreads();
    for (int s = 32; s > 0; s >>= 1) {
        if (tid < s) red[tid] = fmaxf(red[tid], red[tid + s]);
        __syncthreads();
    }
    const float mx = red[0]; __syncthreads();
    float e = (tid < NC) ? __expf(logit - mx) : 0.f;
    red[tid] = e; __syncthreads();
    for (int s = 32; s > 0; s >>= 1) {
        if (tid < s) red[tid] += red[tid + s];
        __syncthreads();
    }
    const float sum = red[0];
    if (tid < NC) out[(size_t)row * NC + tid] = e / sum;
}

// ---------------------------------------------------------------------------
// Launch
// ---------------------------------------------------------------------------
extern "C" void kernel_launch(void* const* d_in, const int* in_sizes, int n_in,
                              void* d_out, int out_size, void* d_ws, size_t ws_size,
                              hipStream_t stream) {
    const float* x1   = (const float*)d_in[0];
    // d_in[1] (x2) is unused by the reference.
    const float* cw0  = (const float*)d_in[2];
    const float* cw1  = (const float*)d_in[3];
    const float* cbv  = (const float*)d_in[4];
    const float* wih0 = (const float*)d_in[5];
    const float* whh0 = (const float*)d_in[6];
    const float* bih0 = (const float*)d_in[7];
    const float* bhh0 = (const float*)d_in[8];
    const float* wih1 = (const float*)d_in[9];
    const float* whh1 = (const float*)d_in[10];
    const float* bih1 = (const float*)d_in[11];
    const float* bhh1 = (const float*)d_in[12];
    const float* fcw  = (const float*)d_in[13];
    const float* fcb  = (const float*)d_in[14];
    float* out = (float*)d_out;

    char* ws = (char*)d_ws;
    float*          lhat  = (float*)(ws + OFF_LHAT);
    float*          b0    = (float*)(ws + OFF_B0);
    float*          b1    = (float*)(ws + OFF_B1);
    unsigned short* wih0p = (unsigned short*)(ws + OFF_WIH0);
    unsigned short* whh0p = (unsigned short*)(ws + OFF_WHH0);
    unsigned short* wih1p = (unsigned short*)(ws + OFF_WIH1);
    unsigned short* whh1p = (unsigned short*)(ws + OFF_WHH1);
    unsigned short* X0    = (unsigned short*)(ws + OFF_X0);
    unsigned short* HS0   = (unsigned short*)(ws + OFF_HS0);
    float*          HLAST = (float*)(ws + OFF_HLAST);

    // Init: Laplacian + bf16 weight packing + combined biases
    build_lhat_kernel<<<1, 256, 0, stream>>>(lhat);
    pad_convert_kernel<<<(512 * 64 + 255) / 256, 256, 0, stream>>>(wih0p, wih0, 512, 45, 64);
    pad_convert_kernel<<<(512 * 128 + 255) / 256, 256, 0, stream>>>(whh0p, whh0, 512, 128, 128);
    pad_convert_kernel<<<(512 * 128 + 255) / 256, 256, 0, stream>>>(wih1p, wih1, 512, 128, 128);
    pad_convert_kernel<<<(512 * 128 + 255) / 256, 256, 0, stream>>>(whh1p, whh1, 512, 128, 128);
    bias_combine_kernel<<<2, 256, 0, stream>>>(b0, bih0, bhh0, 512);
    bias_combine_kernel<<<2, 256, 0, stream>>>(b1, bih1, bhh1, 512);

    // Fused permute + ChebConv + bf16 pack
    cheb_kernel<<<ROWS / 256, 256, 0, stream>>>(x1, cw0, cw1, cbv, lhat, X0);

    // LSTM layer 0 (K_in padded 45->64) and layer 1 (K_in = 128)
    lstm_layer_kernel<64><<<BNM / 16, 256, 0, stream>>>(X0, wih0p, whh0p, b0, HS0, nullptr, TT);
    lstm_layer_kernel<128><<<BNM / 16, 256, 0, stream>>>(HS0, wih1p, whh1p, b1, nullptr, HLAST, TT);

    // Classifier head
    fc_softmax_kernel<<<BNM, 64, 0, stream>>>(HLAST, fcw, fcb, out);
}
